// BitLinear_50766513438886
// MI455X (gfx1250) — 8-bit matrix operand order repaired and checked by reading, compiled, not run on hardware
//
#include <hip/hip_runtime.h>

typedef __attribute__((ext_vector_type(8))) int v8i;

#define K_DIM 2048
#define M_TOK 8192
#define N_OUT 8192
#define QB    127.0f

union AF { v8i v; int2 d[4]; };
union BF { v8i v; int4 q[2]; };

__global__ __launch_bounds__(256) void ln_quant_kernel(const float* __restrict__ x,
                                                       signed char* __restrict__ xq,
                                                       float* __restrict__ eta_out) {
  __shared__ float sred[256];
  const int token = blockIdx.x;
  const int t = threadIdx.x;
  const float* row = x + (size_t)token * K_DIM;

  float v[8];
  float sum = 0.f, sumsq = 0.f;
#pragma unroll
  for (int i = 0; i < 8; ++i) {
    float f = __builtin_nontemporal_load(&row[t + 256 * i]);
    v[i] = f;
    sum += f;
    sumsq += f * f;
  }

  sred[t] = sum; __syncthreads();
  for (int s = 128; s > 0; s >>= 1) { if (t < s) sred[t] += sred[t + s]; __syncthreads(); }
  const float mean = sred[0] * (1.0f / (float)K_DIM);
  __syncthreads();

  sred[t] = sumsq; __syncthreads();
  for (int s = 128; s > 0; s >>= 1) { if (t < s) sred[t] += sred[t + s]; __syncthreads(); }
  const float var = sred[0] * (1.0f / (float)K_DIM) - mean * mean;
  __syncthreads();

  const float rstd = rsqrtf(var + 1e-5f);

  float amax = 0.f;
#pragma unroll
  for (int i = 0; i < 8; ++i) {
    v[i] = (v[i] - mean) * rstd;
    amax = fmaxf(amax, fabsf(v[i]));
  }

  sred[t] = amax; __syncthreads();
  for (int s = 128; s > 0; s >>= 1) { if (t < s) sred[t] = fmaxf(sred[t], sred[t + s]); __syncthreads(); }
  const float eta = fmaxf(sred[0], 1e-5f);
  if (t == 0) eta_out[token] = eta;

  const float qs = QB / eta;
  signed char* qrow = xq + (size_t)token * K_DIM;
#pragma unroll
  for (int i = 0; i < 8; ++i) {
    float q = rintf(fminf(fmaxf(v[i] * qs, -128.f), 127.f));
    qrow[t + 256 * i] = (signed char)(int)q;
  }
}

__global__ __launch_bounds__(256) void absmean_partial_kernel(const float* __restrict__ w,
                                                              float* __restrict__ partials) {
  __shared__ float sred[256];
  const int t = threadIdx.x;
  const size_t base = (size_t)blockIdx.x * 2048;
  float s = 0.f;
#pragma unroll
  for (int i = 0; i < 8; ++i) s += fabsf(w[base + t + 256 * i]);
  sred[t] = s; __syncthreads();
  for (int st = 128; st > 0; st >>= 1) { if (t < st) sred[t] += sred[t + st]; __syncthreads(); }
  if (t == 0) partials[blockIdx.x] = sred[0];
}

__global__ __launch_bounds__(256) void gamma_finalize_kernel(const float* __restrict__ partials,
                                                             float* __restrict__ gamma) {
  __shared__ float sred[256];
  const int t = threadIdx.x;
  float s = 0.f;
  for (int i = t; i < 8192; i += 256) s += partials[i];
  sred[t] = s; __syncthreads();
  for (int st = 128; st > 0; st >>= 1) { if (t < st) sred[t] += sred[t + st]; __syncthreads(); }
  if (t == 0) {
    float g = sred[0] * (1.0f / (float)((size_t)N_OUT * K_DIM));
    gamma[0] = fmaxf(g, 1e-5f);
  }
}

__global__ __launch_bounds__(256) void wquant_kernel(const float* __restrict__ w,
                                                     const float* __restrict__ gamma_p,
                                                     signed char* __restrict__ wq) {
  const float inv_g = 1.0f / gamma_p[0];
  const size_t base = (size_t)blockIdx.x * 2048 + threadIdx.x;
#pragma unroll
  for (int i = 0; i < 8; ++i) {
    float q = rintf(fminf(fmaxf(w[base + 256 * i] * inv_g, -1.f), 1.f));
    wq[base + 256 * i] = (signed char)(int)q;
  }
}

__global__ __launch_bounds__(256) void gemm_iu8_kernel(const signed char* __restrict__ xq,
                                                       const signed char* __restrict__ wq,
                                                       const float* __restrict__ eta,
                                                       const float* __restrict__ gamma_p,
                                                       float* __restrict__ out) {
  const int lane = threadIdx.x & 31;
  const int wave = threadIdx.x >> 5;
  const int half = lane >> 4;
  const int l16  = lane & 15;

  const int wm = wave & 1;
  const int wn = wave >> 1;
  const int tm0 = blockIdx.y * 4 + wm * 2;
  const int tn0 = blockIdx.x * 16 + wn * 4;

  int aOff[2];
  int bOff[4];
#pragma unroll
  for (int i = 0; i < 2; ++i) aOff[i] = ((tm0 + i) * 16 + l16) * K_DIM + half * 8;
#pragma unroll
  for (int j = 0; j < 4; ++j) bOff[j] = ((tn0 + j) * 16 + l16) * K_DIM + half * 8;

  AF a[2][2];
  BF b[2][4];
  v8i acc[2][4] = {};

  auto loadSet = [&](int s, int k) {
    const signed char* ak = xq + k;
    const signed char* bk = wq + k;
#pragma unroll
    for (int i = 0; i < 2; ++i) {
#pragma unroll
      for (int p = 0; p < 4; ++p)
        a[s][i].d[p] = *(const int2*)(ak + aOff[i] + p * 16);
    }
#pragma unroll
    for (int j = 0; j < 4; ++j) {
#pragma unroll
      for (int p = 0; p < 4; ++p)
        ((int2*)&b[s][j])[p] = *(const int2*)(bk + bOff[j] + p * 16);
    }
  };

  auto mmaSet = [&](int s) {
#pragma unroll
    for (int i = 0; i < 2; ++i) {
#pragma unroll
      for (int j = 0; j < 4; ++j) {
        acc[i][j] = __builtin_amdgcn_wmma_i32_16x16x64_iu8(
            true, a[s][i].v, true, b[s][j].v, acc[i][j], false, false);
      }
    }
  };

  loadSet(0, 0);
  int k0 = 0;
#pragma unroll 1
  for (; k0 + 128 < K_DIM; k0 += 128) {
    loadSet(1, k0 + 64);
    mmaSet(0);
    loadSet(0, k0 + 128);
    mmaSet(1);
  }
  loadSet(1, k0 + 64);
  mmaSet(0);
  mmaSet(1);

  const float g = gamma_p[0] * (1.0f / QB);
#pragma unroll
  for (int i = 0; i < 2; ++i) {
#pragma unroll
    for (int r = 0; r < 8; ++r) {
      const int m = (tm0 + i) * 16 + 8 * half + r;
      const float s = g * eta[m];
      float* orow = out + (size_t)m * N_OUT;
#pragma unroll
      for (int j = 0; j < 4; ++j) {
        __builtin_nontemporal_store((float)acc[i][j][r] * s,
                                    &orow[(tn0 + j) * 16 + l16]);
      }
    }
  }
}

extern "C" void kernel_launch(void* const* d_in, const int* in_sizes, int n_in,
                              void* d_out, int out_size, void* d_ws, size_t ws_size,
                              hipStream_t stream) {
  const float* x = (const float*)d_in[0];
  const float* w = (const float*)d_in[1];
  float* out = (float*)d_out;

  char* ws = (char*)d_ws;
  size_t off = 0;
  signed char* xq = (signed char*)(ws + off); off += (size_t)M_TOK * K_DIM;
  signed char* wq = (signed char*)(ws + off); off += (size_t)N_OUT * K_DIM;
  float* eta      = (float*)(ws + off);      off += (size_t)M_TOK * sizeof(float);
  float* gamma    = (float*)(ws + off);      off += 256;
  float* partials = (float*)(ws + off);      off += 8192 * sizeof(float);

  ln_quant_kernel<<<M_TOK, 256, 0, stream>>>(x, xq, eta);

  absmean_partial_kernel<<<8192, 256, 0, stream>>>(w, partials);
  gamma_finalize_kernel<<<1, 256, 0, stream>>>(partials, gamma);

  wquant_kernel<<<8192, 256, 0, stream>>>(w, gamma, wq);

  dim3 grid(N_OUT / 256, M_TOK / 64);
  gemm_iu8_kernel<<<grid, 256, 0, stream>>>(xq, wq, eta, gamma, out);
}
